// SubjectAwareCNNTransformer_18476949307792
// MI455X (gfx1250) — compile-verified, matrix operand order checked by reading, not run on hardware
//
#include <hip/hip_runtime.h>
#include <hip/hip_bf16.h>
#include <math.h>

// ---------------------------------------------------------------------------
// CDNA5 (gfx1250) fused implementation of SubjectAwareCNNTransformer forward.
// wave32, WMMA bf16 16x16x32 with f32 accumulation for all matrix work.
// All WMMA operands (A and B) are fed from per-lane-contiguous storage so the
// fragment loads vectorize to b128 accesses (B matrices are consumed in
// transposed [N][K] form, which has identical lane contiguity to A).
// ---------------------------------------------------------------------------

typedef __attribute__((ext_vector_type(16))) __bf16 v16bf;
typedef __attribute__((ext_vector_type(8)))  float  v8f;

#define DEV __device__ __forceinline__

DEV v8f wmma_bf16(v16bf a, v16bf b, v8f c) {
  // D = A(16x32 bf16) x B(32x16 bf16) + C(16x16 f32)
  return __builtin_amdgcn_wmma_f32_16x16x32_bf16(false, a, false, b, (short)0, c,
                                                 false, false);
}

DEV v8f zero8() {
  v8f z;
#pragma unroll
  for (int i = 0; i < 8; ++i) z[i] = 0.f;
  return z;
}

DEV float gelu_exact(float x) {
  return 0.5f * x * (1.0f + erff(x * 0.70710678118654752f));
}

// K index pattern for 16-bit A/B fragments (ISA 7.12.2):
// VGPR v holds K = {kh + 2v, kh+2v+1} for v<4 and {16+kh+2(v-4), +1} for v>=4,
// kh = 8*(lane>=16).  Per lane this is two contiguous 8-element chunks:
// [kh, kh+8) and [16+kh, 16+kh+8).
DEV int kpat(int v, int kh) { return ((v & 4) << 2) + kh + ((v & 3) << 1); }

// A fragment: 16x32 tile of row-major bf16 matrix, leading dim ld.
DEV v16bf load_a(const __bf16* p, int ld) {
  const int lane = threadIdx.x & 31;
  const int m = lane & 15;
  const int kh = (lane >> 4) << 3;
  v16bf f;
#pragma unroll
  for (int v = 0; v < 8; ++v) {
    const int k = kpat(v, kh);
    f[2 * v]     = p[m * ld + k];
    f[2 * v + 1] = p[m * ld + k + 1];
  }
  return f;
}

// A fragment from f32 source (convert on load).
DEV v16bf load_a_f32(const float* p, int ld) {
  const int lane = threadIdx.x & 31;
  const int m = lane & 15;
  const int kh = (lane >> 4) << 3;
  v16bf f;
#pragma unroll
  for (int v = 0; v < 8; ++v) {
    const int k = kpat(v, kh);
    f[2 * v]     = (__bf16)p[m * ld + k];
    f[2 * v + 1] = (__bf16)p[m * ld + k + 1];
  }
  return f;
}

// A fragment with only K=0..15 valid (zero-padded to 32) — dh=16 attention.
DEV v16bf load_a_pad16(const __bf16* p, int ld) {
  const int lane = threadIdx.x & 31;
  const int m = lane & 15;
  const int kh = (lane >> 4) << 3;
  v16bf f;
#pragma unroll
  for (int v = 0; v < 8; ++v) {
    const int k = kpat(v, kh);
    f[2 * v]     = (k < 16)     ? p[m * ld + k]     : (__bf16)0.f;
    f[2 * v + 1] = (k + 1 < 16) ? p[m * ld + k + 1] : (__bf16)0.f;
  }
  return f;
}

// B fragment consumed from TRANSPOSED storage: memory holds B^T as [N][K]
// row-major with leading dim ld.  B[k][n] = p[n*ld + k]; per-lane contiguous.
DEV v16bf load_bt(const __bf16* p, int ld) {
  const int lane = threadIdx.x & 31;
  const int n = lane & 15;
  const int kh = (lane >> 4) << 3;
  v16bf f;
#pragma unroll
  for (int v = 0; v < 8; ++v) {
    const int k = kpat(v, kh);
    f[2 * v]     = p[n * ld + k];
    f[2 * v + 1] = p[n * ld + k + 1];
  }
  return f;
}

// Same, with only k<16 valid (dh=16 padded to 32) — used for K^T in QK^T.
DEV v16bf load_bt_pad16(const __bf16* p, int ld) {
  const int lane = threadIdx.x & 31;
  const int n = lane & 15;
  const int kh = (lane >> 4) << 3;
  v16bf f;
#pragma unroll
  for (int v = 0; v < 8; ++v) {
    const int k = kpat(v, kh);
    f[2 * v]     = (k < 16)     ? p[n * ld + k]     : (__bf16)0.f;
    f[2 * v + 1] = (k + 1 < 16) ? p[n * ld + k + 1] : (__bf16)0.f;
  }
  return f;
}

// ---------------------------------------------------------------------------
// Kernel 0: weight prep.  PyTorch-layout weights [out][in] are already the
// B^T ([N][K]) layout the fragment loaders want, so transformer weights are a
// plain f32->bf16 cast.  conv2 weights are repacked per kernel tap:
// w2t[kk][co][ci] (A-side row-major per tap).  BN constants are folded.
// ---------------------------------------------------------------------------
__global__ void prep_kernel(
    const float* __restrict__ conv1_b,
    const float* __restrict__ bn1_g, const float* __restrict__ bn1_b,
    const float* __restrict__ bn1_m, const float* __restrict__ bn1_v,
    const float* __restrict__ conv2_w, const float* __restrict__ conv2_b,
    const float* __restrict__ bn2_g, const float* __restrict__ bn2_b,
    const float* __restrict__ bn2_m, const float* __restrict__ bn2_v,
    const float* __restrict__ qkv_w, const float* __restrict__ out_w,
    const float* __restrict__ ff1_w, const float* __restrict__ ff2_w,
    __bf16* __restrict__ w2t,     // [5][64][64]
    __bf16* __restrict__ wqkv,    // [L][384][128]  (cast)
    __bf16* __restrict__ wout,    // [L][128][128]  (cast)
    __bf16* __restrict__ wff1,    // [L][256][128]  (cast)
    __bf16* __restrict__ wff2,    // [L][128][256]  (cast)
    float* __restrict__ bnc)      // s1[64] h1[64] s2[64] h2[64]
{
  const int i0 = blockIdx.x * blockDim.x + threadIdx.x;
  const int stride = gridDim.x * blockDim.x;
  if (i0 < 64) {
    float s1 = bn1_g[i0] * rsqrtf(bn1_v[i0] + 1e-5f);
    bnc[i0]       = s1;
    bnc[64 + i0]  = (conv1_b[i0] - bn1_m[i0]) * s1 + bn1_b[i0];
    float s2 = bn2_g[i0] * rsqrtf(bn2_v[i0] + 1e-5f);
    bnc[128 + i0] = s2;
    bnc[192 + i0] = (conv2_b[i0] - bn2_m[i0]) * s2 + bn2_b[i0];
  }
  for (int i = i0; i < 5 * 64 * 64; i += stride) {
    int kk = i >> 12, r = i & 4095, co = r >> 6, ci = r & 63;
    w2t[i] = (__bf16)conv2_w[(co * 64 + ci) * 5 + kk];
  }
  for (int i = i0; i < 2 * 384 * 128; i += stride) wqkv[i] = (__bf16)qkv_w[i];
  for (int i = i0; i < 2 * 128 * 128; i += stride) wout[i] = (__bf16)out_w[i];
  for (int i = i0; i < 2 * 256 * 128; i += stride) wff1[i] = (__bf16)ff1_w[i];
  for (int i = i0; i < 2 * 128 * 256; i += stride) wff2[i] = (__bf16)ff2_w[i];
}

// ---------------------------------------------------------------------------
// Kernel 1: fused per-sample encoder.  One block per (b,c) sample (7808
// blocks, 8 wave32 waves).  conv1+BN+GELU+maxpool in VALU -> transposed LDS
// plane zt[t][ci]; conv2 computed as 5 shift-GEMMs (one per tap, K=64 each,
// all operands lane-contiguous); BN+GELU+mean epilogue via shfl + LDS
// atomics; 128x64 projection + electrode embedding.
// ---------------------------------------------------------------------------
#define ZROWS 264
__global__ void __launch_bounds__(256) encoder_kernel(
    const float* __restrict__ x,        // [7808,500]
    const float* __restrict__ conv1_w,  // [64,9]
    const float* __restrict__ bnc,      // s1,h1,s2,h2
    const __bf16* __restrict__ w2t,     // [5][64][64]
    const float* __restrict__ proj_w,   // [128,64]
    const float* __restrict__ proj_b,   // [128]
    const float* __restrict__ elec,     // [122,128]
    float* __restrict__ tok)            // [7808,128]
{
  extern __shared__ char smem[];
  __bf16* aw     = (__bf16*)smem;                     // 5*64*64 (40960 B)
  __bf16* zt     = (__bf16*)(smem + 40960);           // 264*64  (33792 B)
  float*  xbuf   = (float*)(smem + 74752);            // 512     (2048 B)
  float*  cw1    = (float*)(smem + 76800);            // 576     (2304 B)
  float*  pooled = (float*)(smem + 79104);            // 64      (256 B)
  float*  lbnc   = (float*)(smem + 79360);            // 256     (1024 B)

  const int tid = threadIdx.x, wave = tid >> 5, lane = tid & 31;
  const int ln_ = lane & 15, mhh = (lane >> 4) << 3;
  const int bc = blockIdx.x;
  const float* xr = x + (size_t)bc * 500;

  for (int i = tid; i < 512; i += 256) {
    int t = i - 4;
    xbuf[i] = (t >= 0 && t < 500) ? xr[t] : 0.f;  // conv1 pad=4
  }
  for (int i = tid; i < 576; i += 256) cw1[i] = conv1_w[i];
  for (int i = tid; i < 256; i += 256) lbnc[i] = bnc[i];
  for (int i = tid; i < 5 * 64 * 64; i += 256) aw[i] = w2t[i];
  for (int i = tid; i < ZROWS * 64; i += 256) zt[i] = (__bf16)0.f;
  if (tid < 64) pooled[tid] = 0.f;
  __syncthreads();

  const float* s1 = lbnc;
  const float* h1 = lbnc + 64;
  const float* s2 = lbnc + 128;
  const float* h2 = lbnc + 192;

  // conv1 (k=9,pad=4) -> BN -> GELU -> maxpool(2): zt[(tp+2)][co], tp<250.
  // Row offset +2 implements the conv2 left padding; rows 0,1 and >=252 stay 0.
  for (int idx = tid; idx < 64 * 250; idx += 256) {
    int co = idx / 250, tp = idx - co * 250, t0 = tp * 2;
    const float* w = cw1 + co * 9;
    float y0 = 0.f, y1 = 0.f;
#pragma unroll
    for (int k = 0; k < 9; ++k) {
      y0 = fmaf(w[k], xbuf[t0 + k], y0);
      y1 = fmaf(w[k], xbuf[t0 + 1 + k], y1);
    }
    float a0 = gelu_exact(y0 * s1[co] + h1[co]);
    float a1 = gelu_exact(y1 * s1[co] + h1[co]);
    zt[(tp + 2) * 64 + co] = (__bf16)fmaxf(a0, a1);
  }
  __syncthreads();

  // conv2: out[co][t] = sum_kk sum_ci W[co][ci][kk] * z[ci][t+kk-2]
  // 4 m-tiles x 16 n-tiles over 8 waves; per tile 5 taps x 2 K-steps = 10 WMMA.
  for (int tile = wave; tile < 64; tile += 8) {
    const int m0 = (tile >> 4) * 16, n0 = (tile & 15) * 16;
    v8f acc = zero8();
#pragma unroll
    for (int kk = 0; kk < 5; ++kk) {
      const __bf16* wt = aw + (kk << 12) + m0 * 64;  // [co][ci] tap slab
      const __bf16* zr = zt + (n0 + kk) * 64;        // shifted z rows (B^T)
      acc = wmma_bf16(load_a(wt, 64),      load_bt(zr, 64),      acc);
      acc = wmma_bf16(load_a(wt + 32, 64), load_bt(zr + 32, 64), acc);
    }
    const int n = n0 + ln_;
#pragma unroll
    for (int j = 0; j < 8; ++j) {
      const int m = m0 + mhh + j;
      float v = gelu_exact(acc[j] * s2[m] + h2[m]);
      v = (n < 250) ? v : 0.f;  // mask padded time steps before the mean
#pragma unroll
      for (int off = 1; off < 16; off <<= 1) v += __shfl_xor(v, off, 32);
      if (ln_ == 0) atomicAdd(&pooled[m], v);
    }
  }
  __syncthreads();
  if (tid < 64) pooled[tid] *= (1.f / 250.f);
  __syncthreads();

  // projection 64->128 + electrode embedding
  if (tid < 128) {
    const int c = bc % 122;
    float s = proj_b[tid];
    const float* pw = proj_w + tid * 64;
#pragma unroll 8
    for (int co = 0; co < 64; ++co) s = fmaf(pw[co], pooled[co], s);
    tok[(size_t)bc * 128 + tid] = s + elec[c * 128 + tid];
  }
}

// ---------------------------------------------------------------------------
// Kernel 2: one full transformer encoder layer per block (one batch element
// per block, seq 122 padded to 128, D=128, H=8, dh=16, FF=256).
// Q,K stored [seq][d]; V stored transposed [d][seq] so every B fragment in
// the layer loads contiguously.  288 KB LDS; heads processed serially.
// ---------------------------------------------------------------------------
__global__ void __launch_bounds__(256) xform_kernel(
    float* __restrict__ tok,            // [64*122,128]
    const __bf16* __restrict__ wqkv,    // [L][384][128]
    const __bf16* __restrict__ wout,    // [L][128][128]
    const __bf16* __restrict__ wff1,    // [L][256][128]
    const __bf16* __restrict__ wff2,    // [L][128][256]
    const float* __restrict__ qkv_b, const float* __restrict__ out_b,
    const float* __restrict__ ln1_g, const float* __restrict__ ln1_b,
    const float* __restrict__ ff1_b, const float* __restrict__ ff2_b,
    const float* __restrict__ ln2_g, const float* __restrict__ ln2_b,
    int layer)
{
  extern __shared__ char smem[];
  float*  tokf = (float*)smem;                    // 128*128 f32  (65536 B)
  float*  sbuf = (float*)(smem + 65536);          // 128*128 f32  (65536 B)
  __bf16* tokb = (__bf16*)(smem + 131072);        // 128*128 bf16 (32768 B)
  __bf16* ctxb = (__bf16*)(smem + 163840);        // 128*128 bf16 (32768 B)
  __bf16* qkb  = (__bf16*)(smem + 196608);        // 128*256 bf16 (65536 B)
  __bf16* vt   = (__bf16*)(smem + 262144);        // 128*128 bf16 (32768 B)

  const int tid = threadIdx.x, wave = tid >> 5, lane = tid & 31;
  const int ln_ = lane & 15, mhh = (lane >> 4) << 3;
  float* tokg = tok + (size_t)blockIdx.x * 122 * 128;

  const __bf16* Wq = wqkv + layer * 384 * 128;
  const __bf16* Wo = wout + layer * 128 * 128;
  const __bf16* W1 = wff1 + layer * 256 * 128;
  const __bf16* W2 = wff2 + layer * 128 * 256;
  const float* bq = qkv_b + layer * 384;
  const float* bo = out_b + layer * 128;
  const float* bf1 = ff1_b + layer * 256;
  const float* bf2 = ff2_b + layer * 128;
  const float* g1 = ln1_g + layer * 128, *e1 = ln1_b + layer * 128;
  const float* g2 = ln2_g + layer * 128, *e2 = ln2_b + layer * 128;

  for (int i = tid; i < 128 * 128; i += 256) {
    const int r = i >> 7;
    const float v = (r < 122) ? tokg[i] : 0.f;
    tokf[i] = v;
    tokb[i] = (__bf16)v;
  }
  __syncthreads();

  // ---- QKV GEMM: [128x128] x [128x384]; V epilogue stores transposed ----
  for (int tile = wave; tile < 192; tile += 8) {
    const int m0 = (tile / 24) * 16, n0 = (tile % 24) * 16;
    v8f acc = zero8();
    for (int k0 = 0; k0 < 128; k0 += 32)
      acc = wmma_bf16(load_a(tokb + m0 * 128 + k0, 128),
                      load_bt(Wq + n0 * 128 + k0, 128), acc);
    const int n = n0 + ln_;
    const float bias = bq[n];
    if (n0 < 256) {  // Q,K -> row-major [seq][256]
#pragma unroll
      for (int j = 0; j < 8; ++j)
        qkb[(m0 + mhh + j) * 256 + n] = (__bf16)(acc[j] + bias);
    } else {         // V -> transposed [d][seq]
#pragma unroll
      for (int j = 0; j < 8; ++j)
        vt[(n - 256) * 128 + (m0 + mhh + j)] = (__bf16)(acc[j] + bias);
    }
  }
  __syncthreads();

  // ---- attention, head-serial ----
  for (int h = 0; h < 8; ++h) {
    const int qo = h * 16, ko = 128 + h * 16;
    // scores S = Q K^T / 4   (K-dim 16 zero-padded to 32)
    for (int tile = wave; tile < 64; tile += 8) {
      const int m0 = (tile >> 3) * 16, n0 = (tile & 7) * 16;
      v8f acc = zero8();
      acc = wmma_bf16(load_a_pad16(qkb + m0 * 256 + qo, 256),
                      load_bt_pad16(qkb + n0 * 256 + ko, 256), acc);
      const int n = n0 + ln_;
#pragma unroll
      for (int j = 0; j < 8; ++j)
        sbuf[(m0 + mhh + j) * 128 + n] = (n < 122) ? acc[j] * 0.25f : -1e30f;
    }
    __syncthreads();
    // softmax, one row per thread (in-place f32)
    if (tid < 128) {
      float* row = sbuf + tid * 128;
      float mx = -1e30f;
      for (int n2 = 0; n2 < 128; ++n2) mx = fmaxf(mx, row[n2]);
      float s = 0.f;
      for (int n2 = 0; n2 < 128; ++n2) { float e = __expf(row[n2] - mx); row[n2] = e; s += e; }
      const float inv = 1.f / s;
      for (int n2 = 0; n2 < 128; ++n2) row[n2] *= inv;
    }
    __syncthreads();
    // ctx = P V : M=128 N=16 K=128 (seq); B^T is exactly vt[d][seq]
    {
      const int m0 = wave * 16;
      v8f acc = zero8();
      for (int k0 = 0; k0 < 128; k0 += 32)
        acc = wmma_bf16(load_a_f32(sbuf + m0 * 128 + k0, 128),
                        load_bt(vt + (h * 16) * 128 + k0, 128), acc);
#pragma unroll
      for (int j = 0; j < 8; ++j)
        ctxb[(m0 + mhh + j) * 128 + h * 16 + ln_] = (__bf16)acc[j];
    }
    __syncthreads();
  }

  // ---- out projection + residual ----
  for (int tile = wave; tile < 64; tile += 8) {
    const int m0 = (tile >> 3) * 16, n0 = (tile & 7) * 16;
    v8f acc = zero8();
    for (int k0 = 0; k0 < 128; k0 += 32)
      acc = wmma_bf16(load_a(ctxb + m0 * 128 + k0, 128),
                      load_bt(Wo + n0 * 128 + k0, 128), acc);
    const int n = n0 + ln_;
    const float bias = bo[n];
#pragma unroll
    for (int j = 0; j < 8; ++j)
      tokf[(m0 + mhh + j) * 128 + n] += acc[j] + bias;
  }
  __syncthreads();

  // ---- LN1 (post-norm), refresh bf16 copy ----
  if (tid < 128) {
    float* row = tokf + tid * 128;
    float mu = 0.f;
    for (int c = 0; c < 128; ++c) mu += row[c];
    mu *= (1.f / 128.f);
    float var = 0.f;
    for (int c = 0; c < 128; ++c) { float d = row[c] - mu; var = fmaf(d, d, var); }
    const float inv = rsqrtf(var * (1.f / 128.f) + 1e-5f);
    for (int c = 0; c < 128; ++c) {
      const float y = (row[c] - mu) * inv * g1[c] + e1[c];
      row[c] = y;
      tokb[tid * 128 + c] = (__bf16)y;
    }
  }
  __syncthreads();

  // ---- FF1 + GELU (reuse qkb as [128][256] bf16) ----
  __bf16* ffb = qkb;
  for (int tile = wave; tile < 128; tile += 8) {
    const int m0 = (tile >> 4) * 16, n0 = (tile & 15) * 16;
    v8f acc = zero8();
    for (int k0 = 0; k0 < 128; k0 += 32)
      acc = wmma_bf16(load_a(tokb + m0 * 128 + k0, 128),
                      load_bt(W1 + n0 * 128 + k0, 128), acc);
    const int n = n0 + ln_;
    const float bias = bf1[n];
#pragma unroll
    for (int j = 0; j < 8; ++j)
      ffb[(m0 + mhh + j) * 256 + n] = (__bf16)gelu_exact(acc[j] + bias);
  }
  __syncthreads();

  // ---- FF2 + residual ----
  for (int tile = wave; tile < 64; tile += 8) {
    const int m0 = (tile >> 3) * 16, n0 = (tile & 7) * 16;
    v8f acc = zero8();
    for (int k0 = 0; k0 < 256; k0 += 32)
      acc = wmma_bf16(load_a(ffb + m0 * 256 + k0, 256),
                      load_bt(W2 + n0 * 256 + k0, 256), acc);
    const int n = n0 + ln_;
    const float bias = bf2[n];
#pragma unroll
    for (int j = 0; j < 8; ++j)
      tokf[(m0 + mhh + j) * 128 + n] += acc[j] + bias;
  }
  __syncthreads();

  // ---- LN2 + writeback ----
  if (tid < 128) {
    float* row = tokf + tid * 128;
    float mu = 0.f;
    for (int c = 0; c < 128; ++c) mu += row[c];
    mu *= (1.f / 128.f);
    float var = 0.f;
    for (int c = 0; c < 128; ++c) { float d = row[c] - mu; var = fmaf(d, d, var); }
    const float inv = rsqrtf(var * (1.f / 128.f) + 1e-5f);
    for (int c = 0; c < 128; ++c) {
      const float y = (row[c] - mu) * inv * g2[c] + e2[c];
      if (tid < 122) tokg[tid * 128 + c] = y;
    }
  }
}

// ---------------------------------------------------------------------------
// Kernel 3: sequence mean-pool + per-subject head (tiny).
// ---------------------------------------------------------------------------
__global__ void __launch_bounds__(128) head_kernel(
    const float* __restrict__ tok, const int* __restrict__ sid,
    const float* __restrict__ hw, const float* __restrict__ hb,
    float* __restrict__ out)
{
  __shared__ float pooled[128];
  const int b = blockIdx.x, tid = threadIdx.x;
  if (tid < 128) {
    float s = 0.f;
    const float* base = tok + (size_t)b * 122 * 128 + tid;
    for (int c = 0; c < 122; ++c) s += base[c * 128];
    pooled[tid] = s * (1.f / 122.f);
  }
  __syncthreads();
  if (tid < 4) {
    const int sj = sid[b];
    const float* w = hw + (sj * 4 + tid) * 128;
    float s = hb[sj * 4 + tid];
#pragma unroll 8
    for (int d = 0; d < 128; ++d) s = fmaf(w[d], pooled[d], s);
    out[b * 4 + tid] = s;
  }
}

// ---------------------------------------------------------------------------
extern "C" void kernel_launch(void* const* d_in, const int* in_sizes, int n_in,
                              void* d_out, int out_size, void* d_ws, size_t ws_size,
                              hipStream_t stream) {
  (void)in_sizes; (void)n_in; (void)out_size; (void)ws_size;
  const float* x        = (const float*)d_in[0];
  const int*   sid      = (const int*)d_in[1];
  const float* conv1_w  = (const float*)d_in[2];
  const float* conv1_b  = (const float*)d_in[3];
  const float* bn1_g    = (const float*)d_in[4];
  const float* bn1_b    = (const float*)d_in[5];
  const float* bn1_m    = (const float*)d_in[6];
  const float* bn1_v    = (const float*)d_in[7];
  const float* conv2_w  = (const float*)d_in[8];
  const float* conv2_b  = (const float*)d_in[9];
  const float* bn2_g    = (const float*)d_in[10];
  const float* bn2_b    = (const float*)d_in[11];
  const float* bn2_m    = (const float*)d_in[12];
  const float* bn2_v    = (const float*)d_in[13];
  const float* proj_w   = (const float*)d_in[14];
  const float* proj_b   = (const float*)d_in[15];
  const float* elec     = (const float*)d_in[16];
  const float* qkv_w    = (const float*)d_in[17];
  const float* qkv_b    = (const float*)d_in[18];
  const float* out_w    = (const float*)d_in[19];
  const float* out_b    = (const float*)d_in[20];
  const float* ln1_g    = (const float*)d_in[21];
  const float* ln1_b    = (const float*)d_in[22];
  const float* ff1_w    = (const float*)d_in[23];
  const float* ff1_b    = (const float*)d_in[24];
  const float* ff2_w    = (const float*)d_in[25];
  const float* ff2_b    = (const float*)d_in[26];
  const float* ln2_g    = (const float*)d_in[27];
  const float* ln2_b    = (const float*)d_in[28];
  const float* heads_w  = (const float*)d_in[29];
  const float* heads_b  = (const float*)d_in[30];

  // workspace carve (256-byte aligned slabs)
  char* ws = (char*)d_ws;
  size_t off = 0;
  auto take = [&](size_t bytes) -> char* {
    char* p = ws + off;
    off += (bytes + 255) & ~(size_t)255;
    return p;
  };
  float*  tokws = (float*)take((size_t)7808 * 128 * sizeof(float));
  __bf16* w2t   = (__bf16*)take((size_t)5 * 64 * 64 * 2);
  __bf16* wqkvb = (__bf16*)take((size_t)2 * 384 * 128 * 2);
  __bf16* woutb = (__bf16*)take((size_t)2 * 128 * 128 * 2);
  __bf16* wff1b = (__bf16*)take((size_t)2 * 256 * 128 * 2);
  __bf16* wff2b = (__bf16*)take((size_t)2 * 128 * 256 * 2);
  float*  bnc   = (float*)take(256 * sizeof(float));

  prep_kernel<<<128, 256, 0, stream>>>(
      conv1_b, bn1_g, bn1_b, bn1_m, bn1_v,
      conv2_w, conv2_b, bn2_g, bn2_b, bn2_m, bn2_v,
      qkv_w, out_w, ff1_w, ff2_w,
      w2t, wqkvb, woutb, wff1b, wff2b, bnc);

  encoder_kernel<<<7808, 256, 80384, stream>>>(
      x, conv1_w, bnc, w2t, proj_w, proj_b, elec, tokws);

  for (int l = 0; l < 2; ++l)
    xform_kernel<<<64, 256, 294912, stream>>>(
        tokws, wqkvb, woutb, wff1b, wff2b,
        qkv_b, out_b, ln1_g, ln1_b, ff1_b, ff2_b, ln2_g, ln2_b, l);

  head_kernel<<<64, 128, 0, stream>>>(tokws, sid, heads_w, heads_b, (float*)d_out);
}
